// MLA_41764261987034
// MI455X (gfx1250) — matrix operand order repaired and checked by reading, not compiled, not run on hardware
//
#include <hip/hip_runtime.h>
#include <hip/hip_bf16.h>

// ---------------- MLA dims (match reference) ----------------
#define S_SEQ 2048
#define H_DIM 2048
#define NH 16
#define DN 128
#define DR 64
#define DV 128
#define QL 1536
#define KL 512
#define DQK 192            // DN + DR
#define FUSED_N 2112       // QL + KL + DR
#define QB_N 3072          // NH*(DN+DR)
#define KVB_N 4096         // NH*(DN+DV)

typedef __bf16 bf16_t;
typedef __attribute__((ext_vector_type(16))) __bf16 v16bf;
typedef __attribute__((ext_vector_type(8)))  float  v8f;

union FragU { v16bf v; unsigned u[8]; };

// A-matrix 16x32 bf16 lane layout (ISA 7.12.2): half = lane>>4,
// VGPR v<4 : K = half*8 + 2v ; VGPR v>=4 : K = 16 + half*8 + 2(v-4)
__device__ inline v16bf load_frag_a(const bf16_t* rowbase, int half) {
  FragU f;
  const unsigned* p = reinterpret_cast<const unsigned*>(rowbase);
  int o = half * 4;
#pragma unroll
  for (int i = 0; i < 4; ++i) f.u[i] = p[o + i];
#pragma unroll
  for (int i = 0; i < 4; ++i) f.u[4 + i] = p[8 + o + i];
  return f.v;
}

// B-matrix 32x16 bf16: lane holds column n=lane&15, K same order as A (half*8 + 0..7, 16 + half*8 + 0..7)
__device__ inline v16bf load_frag_b(const bf16_t* colbase, int half) {
  FragU f;
  const unsigned* p = reinterpret_cast<const unsigned*>(colbase);
#pragma unroll
  for (int i = 0; i < 4; ++i) f.u[i] = p[half * 4 + i];
#pragma unroll
  for (int i = 0; i < 4; ++i) f.u[4 + i] = p[8 + half * 4 + i];
  return f.v;
}

__device__ inline v8f wmma_bf16(v16bf a, v16bf b, v8f c) {
  return __builtin_amdgcn_wmma_f32_16x16x32_bf16(false, a, false, b,
                                                 (short)0, c, false, false);
}

// CDNA5 async copy: 16B global -> LDS, tracked by ASYNCcnt (ISA 10.x / 15.18.3)
__device__ inline void async_copy_b128(bf16_t* lds_dst, const bf16_t* g_src) {
  unsigned loff = (unsigned)(unsigned long long)(uintptr_t)lds_dst; // LDS byte offset
  unsigned long long ga = (unsigned long long)(uintptr_t)g_src;
  asm volatile("global_load_async_to_lds_b128 %0, %1, off"
               :: "v"(loff), "v"(ga) : "memory");
}
__device__ inline void wait_async0() {
  asm volatile("s_wait_asynccnt 0x0" ::: "memory");
}

// ---------------- fp32 -> bf16 convert ----------------
__global__ void f32_to_bf16_kernel(const float* __restrict__ in,
                                   bf16_t* __restrict__ out, size_t n) {
  size_t i = (size_t)blockIdx.x * blockDim.x + threadIdx.x;
  size_t stride = (size_t)gridDim.x * blockDim.x;
  for (; i < n; i += stride) out[i] = (bf16_t)in[i];
}

// ---------------- fp32 [K][N] -> bf16 [N][K] LDS-tiled transpose ----------------
__global__ __launch_bounds__(256)
void transpose_f32_bf16_kernel(const float* __restrict__ in, bf16_t* __restrict__ out,
                               int K, int N) {
  __shared__ float tile[32][33];
  int k0 = blockIdx.y * 32, n0 = blockIdx.x * 32;
  int tx = threadIdx.x & 31, ty = threadIdx.x >> 5;   // ty in 0..7
#pragma unroll
  for (int i = 0; i < 32; i += 8)
    tile[ty + i][tx] = in[(size_t)(k0 + ty + i) * N + n0 + tx];
  __syncthreads();
#pragma unroll
  for (int i = 0; i < 32; i += 8)
    out[(size_t)(n0 + ty + i) * K + k0 + tx] = (bf16_t)tile[tx][ty + i];
}

// ---------------- RMSNorm (one wave per row, wave32) ----------------
__global__ __launch_bounds__(256)
void rmsnorm_bf16_kernel(const float* __restrict__ in, const float* __restrict__ gamma,
                         bf16_t* __restrict__ out, int cols, int in_stride, int in_off) {
  int wid = threadIdx.x >> 5, lane = threadIdx.x & 31;
  int row = blockIdx.x * 8 + wid;
  const float* x = in + (size_t)row * in_stride + in_off;
  float ss = 0.f;
  for (int c = lane; c < cols; c += 32) { float v = x[c]; ss += v * v; }
#pragma unroll
  for (int m = 16; m >= 1; m >>= 1) ss += __shfl_xor(ss, m, 32);
  float r = rsqrtf(ss / (float)cols + 1e-6f);
  bf16_t* o = out + (size_t)row * cols;
  for (int c = lane; c < cols; c += 32) o[c] = (bf16_t)(x[c] * gamma[c] * r);
}

// ---------------- bf16 WMMA GEMM: C[M,N] = A[M,K] * Bt[N,K]^T ----------------
// Block tile 128x64, K-step 64; async double-buffered LDS; 8 waves in 4(M)x2(N);
// wave tile 32x32 -> 2x2 accumulators, 8 WMMAs per staging round.
__global__ __launch_bounds__(256)
void gemm_bf16_kernel(const bf16_t* __restrict__ A, const bf16_t* __restrict__ Bt,
                      float* __restrict__ C, int M, int N, int K) {
  __shared__ __align__(16) bf16_t As[2][128 * 64];   // [row][k]  16KB x2
  __shared__ __align__(16) bf16_t Bs[2][64 * 64];    // [n][k]     8KB x2
  int tid = threadIdx.x;
  int wid = tid >> 5, lane = tid & 31;
  int wm = wid & 3, wn = wid >> 2;
  int bm = blockIdx.y * 128, bn = blockIdx.x * 64;
  int half = lane >> 4, l16 = lane & 15;

  v8f acc[2][2];
#pragma unroll
  for (int i = 0; i < 2; ++i)
#pragma unroll
    for (int j = 0; j < 2; ++j) acc[i][j] = (v8f){};

  auto stage = [&](int k0, int buf) {
    // A tile: 128 rows x 64 k = 1024 x 16B chunks, 4 per thread
#pragma unroll
    for (int i = 0; i < 4; ++i) {
      int c = tid + i * 256;
      int row = c >> 3, sub = c & 7;
      async_copy_b128(&As[buf][row * 64 + sub * 8],
                      A + (size_t)(bm + row) * K + k0 + sub * 8);
    }
    // B tile: 64 n-rows x 64 k = 512 x 16B chunks, 2 per thread
#pragma unroll
    for (int i = 0; i < 2; ++i) {
      int c = tid + i * 256;
      int row = c >> 3, sub = c & 7;
      async_copy_b128(&Bs[buf][row * 64 + sub * 8],
                      Bt + (size_t)(bn + row) * K + k0 + sub * 8);
    }
  };

  int buf = 0;
  stage(0, 0);
  wait_async0();
  __syncthreads();

  for (int k0 = 0; k0 < K; k0 += 64) {
    if (k0 + 64 < K) stage(k0 + 64, buf ^ 1);
    const bf16_t* as = As[buf];
    const bf16_t* bs = Bs[buf];
#pragma unroll
    for (int ks = 0; ks < 64; ks += 32) {
      v16bf a0 = load_frag_a(as + (wm * 32 + l16) * 64 + ks, half);
      v16bf a1 = load_frag_a(as + (wm * 32 + 16 + l16) * 64 + ks, half);
      v16bf b0 = load_frag_b(bs + (wn * 32 + l16) * 64 + ks, half);
      v16bf b1 = load_frag_b(bs + (wn * 32 + 16 + l16) * 64 + ks, half);
      acc[0][0] = wmma_bf16(a0, b0, acc[0][0]);
      acc[0][1] = wmma_bf16(a0, b1, acc[0][1]);
      acc[1][0] = wmma_bf16(a1, b0, acc[1][0]);
      acc[1][1] = wmma_bf16(a1, b1, acc[1][1]);
    }
    wait_async0();
    __syncthreads();
    buf ^= 1;
  }

  // C/D layout: VGPR r, lanes0-15 -> M=r, lanes16-31 -> M=8+r; N=lane&15
#pragma unroll
  for (int i = 0; i < 2; ++i) {
    int row0 = bm + wm * 32 + i * 16 + half * 8;
#pragma unroll
    for (int j = 0; j < 2; ++j) {
      int col0 = bn + wn * 32 + j * 16 + l16;
#pragma unroll
      for (int r = 0; r < 8; ++r)
        C[(size_t)(row0 + r) * N + col0] = acc[i][j][r];
    }
  }
}

// ---------------- RoPE + assemble q_full / k_full (bf16) ----------------
__global__ void rope_assemble_kernel(const float* __restrict__ q,
                                     const float* __restrict__ kv,
                                     const float* __restrict__ fused,
                                     bf16_t* __restrict__ qfull,
                                     bf16_t* __restrict__ kfull) {
  int idx = blockIdx.x * blockDim.x + threadIdx.x;
  int d = idx % DQK;
  int h = (idx / DQK) % NH;
  int s = idx / (DQK * NH);
  const float scale = 0.07216878364870323f; // 1/sqrt(192), folded into q

  float qv, kvv;
  const float* qrow = q + (size_t)s * QB_N + h * DQK;
  if (d < DN) {
    qv  = qrow[d];
    kvv = kv[(size_t)s * KVB_N + h * (DN + DV) + d];
  } else {
    int dd = d - DN;                 // 0..63
    int i  = dd & 31;                // rotary freq index
    float inv = powf(10000.f, -(2.f * (float)i) / 64.f);
    float ang = (float)s * inv;
    float c = __cosf(ang), sn = __sinf(ang);
    float x  = qrow[DN + dd];
    float xr = (dd < 32) ? -qrow[DN + dd + 32] : qrow[DN + dd - 32];
    qv = x * c + xr * sn;
    const float* kp = fused + (size_t)s * FUSED_N + (QL + KL);
    float y  = kp[dd];
    float yr = (dd < 32) ? -kp[dd + 32] : kp[dd - 32];
    kvv = y * c + yr * sn;
  }
  qfull[idx] = (bf16_t)(qv * scale);
  kfull[idx] = (bf16_t)kvv;
}

// v transposed per head: vtg[h][d][s]  (async-stageable rows over s)
__global__ void extract_v_t_kernel(const float* __restrict__ kv, bf16_t* __restrict__ vtg) {
  int idx = blockIdx.x * blockDim.x + threadIdx.x;
  int s = idx & (S_SEQ - 1);
  int d = (idx >> 11) & 127;
  int h = idx >> 18;
  vtg[idx] = (bf16_t)kv[(size_t)s * KVB_N + h * (DN + DV) + DN + d];
}

// ---------------- Flash attention with WMMA + async staging ----------------
// grid: (S/64, NH); 128 threads = 4 waves; wave handles 16 queries.
__global__ __launch_bounds__(128)
void mla_attn_kernel(const bf16_t* __restrict__ qf, const bf16_t* __restrict__ kf,
                     const bf16_t* __restrict__ vtg, bf16_t* __restrict__ attn) {
  __shared__ __align__(16) bf16_t kt[32 * DQK];      // [key][d]      12 KB
  __shared__ __align__(16) bf16_t vt[DV * 32];       // [d][key]       8 KB
  __shared__ __align__(16) bf16_t pt[4][16 * 32];    // per-wave probs 4 KB

  int tid = threadIdx.x;
  int wid = tid >> 5, lane = tid & 31;
  int half = lane >> 4, l16 = lane & 15;
  int h  = blockIdx.y;
  int q0 = blockIdx.x * 64;
  int qrow = q0 + wid * 16 + l16;

  // Q fragments (scale folded in): 6 K-steps of 32 over DQK=192
  v16bf qfrag[6];
  const bf16_t* qbase = qf + ((size_t)qrow * NH + h) * DQK;
#pragma unroll
  for (int kc = 0; kc < 6; ++kc) qfrag[kc] = load_frag_a(qbase + kc * 32, half);

  v8f acc[8];
#pragma unroll
  for (int c = 0; c < 8; ++c) acc[c] = (v8f){};
  float run_max[8], run_sum[8];
#pragma unroll
  for (int r = 0; r < 8; ++r) { run_max[r] = -1e30f; run_sum[r] = 0.f; }

  int nkb = (q0 + 64) / 32;            // causal: key blocks of 32
  for (int kb = 0; kb < nkb; ++kb) {
    int kbase = kb * 32;
    // async stage K tile: 32 keys x 192 = 768 x 16B chunks, 6 per thread
#pragma unroll
    for (int i = 0; i < 6; ++i) {
      int c = tid + i * 128;
      int key = c / 24, sub = c % 24;
      async_copy_b128(kt + key * DQK + sub * 8,
                      kf + ((size_t)(kbase + key) * NH + h) * DQK + sub * 8);
    }
    // async stage V tile (pre-transposed): 128 d-rows x 32 keys = 512 chunks, 4/thread
#pragma unroll
    for (int i = 0; i < 4; ++i) {
      int c = tid + i * 128;
      int d = c >> 2, sub = c & 3;
      async_copy_b128(vt + d * 32 + sub * 8,
                      vtg + ((size_t)h * DV + d) * S_SEQ + kbase + sub * 8);
    }
    wait_async0();
    __syncthreads();

    // scores: 16 queries x 32 keys
    v8f c0 = {}; v8f c1 = {};
#pragma unroll
    for (int kc = 0; kc < 6; ++kc) {
      v16bf b0 = load_frag_b(kt + l16 * DQK + kc * 32, half);
      v16bf b1 = load_frag_b(kt + (16 + l16) * DQK + kc * 32, half);
      c0 = wmma_bf16(qfrag[kc], b0, c0);
      c1 = wmma_bf16(qfrag[kc], b1, c1);
    }

    // causal mask + online softmax (rows = vgpr r + 8*half, col = l16)
    int qg = q0 + wid * 16 + half * 8;
    float p0[8], p1[8], pm[8];
#pragma unroll
    for (int r = 0; r < 8; ++r) {
      int qi = qg + r;
      p0[r] = (kbase + l16      <= qi) ? c0[r] : -1e30f;
      p1[r] = (kbase + 16 + l16 <= qi) ? c1[r] : -1e30f;
      pm[r] = fmaxf(p0[r], p1[r]);
    }
#pragma unroll
    for (int m = 8; m >= 1; m >>= 1)
#pragma unroll
      for (int r = 0; r < 8; ++r) pm[r] = fmaxf(pm[r], __shfl_xor(pm[r], m, 32));

    float rs[8];
#pragma unroll
    for (int r = 0; r < 8; ++r) {
      float nm = fmaxf(run_max[r], pm[r]);
      p0[r] = __expf(p0[r] - nm);
      p1[r] = __expf(p1[r] - nm);
      rs[r] = p0[r] + p1[r];
      float f = __expf(run_max[r] - nm);
      run_sum[r] *= f;
      run_max[r] = nm;
      pm[r] = f;                        // rescale factor
    }
#pragma unroll
    for (int m = 8; m >= 1; m >>= 1)
#pragma unroll
      for (int r = 0; r < 8; ++r) rs[r] += __shfl_xor(rs[r], m, 32);
#pragma unroll
    for (int r = 0; r < 8; ++r) run_sum[r] += rs[r];
#pragma unroll
    for (int c = 0; c < 8; ++c)
#pragma unroll
      for (int r = 0; r < 8; ++r) acc[c][r] *= pm[r];

    // transpose probs C-layout -> A-layout through per-wave LDS
#pragma unroll
    for (int r = 0; r < 8; ++r) {
      pt[wid][(half * 8 + r) * 32 + l16]      = (bf16_t)p0[r];
      pt[wid][(half * 8 + r) * 32 + 16 + l16] = (bf16_t)p1[r];
    }
    asm volatile("s_wait_dscnt 0x0" ::: "memory");
    v16bf pfrag = load_frag_a(&pt[wid][l16 * 32], half);

    // P(16x32) x V(32x128): 8 output chunks of 16
#pragma unroll
    for (int c = 0; c < 8; ++c) {
      v16bf vfrag = load_frag_b(vt + (c * 16 + l16) * 32, half);
      acc[c] = wmma_bf16(pfrag, vfrag, acc[c]);
    }
    __syncthreads();
  }

  int qg = q0 + wid * 16 + half * 8;
#pragma unroll
  for (int c = 0; c < 8; ++c)
#pragma unroll
    for (int r = 0; r < 8; ++r)
      attn[(size_t)(qg + r) * (NH * DV) + h * DV + c * 16 + l16] =
          (bf16_t)(acc[c][r] / run_sum[r]);
}

// ---------------- host launch ----------------
extern "C" void kernel_launch(void* const* d_in, const int* in_sizes, int n_in,
                              void* d_out, int out_size, void* d_ws, size_t ws_size,
                              hipStream_t stream) {
  const float* hidden   = (const float*)d_in[0];
  const float* w_kv_a   = (const float*)d_in[1];
  const float* q_gamma  = (const float*)d_in[2];
  const float* w_qb     = (const float*)d_in[3];
  const float* kv_gamma = (const float*)d_in[4];
  const float* w_kvb    = (const float*)d_in[5];
  const float* w_o      = (const float*)d_in[6];
  float* out = (float*)d_out;

  char* ws = (char*)d_ws;
  size_t off = 0;
  auto alloc = [&](size_t bytes) -> void* {
    void* p = ws + off;
    off = (off + bytes + 255) & ~(size_t)255;
    return p;
  };
  bf16_t* hidden_bf = (bf16_t*)alloc((size_t)S_SEQ * H_DIM * 2);
  bf16_t* wkva_t    = (bf16_t*)alloc((size_t)H_DIM * FUSED_N * 2);   // [N][K]
  float*  fused     = (float*) alloc((size_t)S_SEQ * FUSED_N * 4);
  bf16_t* qn_bf     = (bf16_t*)alloc((size_t)S_SEQ * QL * 2);
  bf16_t* wqb_t     = (bf16_t*)alloc((size_t)QL * QB_N * 2);         // [N][K]
  float*  q_f       = (float*) alloc((size_t)S_SEQ * QB_N * 4);
  bf16_t* kvn_bf    = (bf16_t*)alloc((size_t)S_SEQ * KL * 2);
  bf16_t* wkvb_t    = (bf16_t*)alloc((size_t)KL * KVB_N * 2);        // [N][K]
  float*  kv_f      = (float*) alloc((size_t)S_SEQ * KVB_N * 4);
  bf16_t* qfull     = (bf16_t*)alloc((size_t)S_SEQ * NH * DQK * 2);
  bf16_t* kfull     = (bf16_t*)alloc((size_t)S_SEQ * NH * DQK * 2);
  bf16_t* vtg       = (bf16_t*)alloc((size_t)NH * DV * S_SEQ * 2);   // [h][d][s]
  bf16_t* attn_bf   = (bf16_t*)alloc((size_t)S_SEQ * NH * DV * 2);
  bf16_t* wo_t      = (bf16_t*)alloc((size_t)(NH * DV) * H_DIM * 2); // [N][K]

  // converts + weight transposes (fp32 [K][N] -> bf16 [N][K])
  f32_to_bf16_kernel<<<2048, 256, 0, stream>>>(hidden, hidden_bf, (size_t)S_SEQ * H_DIM);
  transpose_f32_bf16_kernel<<<dim3(FUSED_N / 32, H_DIM / 32), 256, 0, stream>>>(
      w_kv_a, wkva_t, H_DIM, FUSED_N);
  transpose_f32_bf16_kernel<<<dim3(QB_N / 32, QL / 32), 256, 0, stream>>>(
      w_qb, wqb_t, QL, QB_N);
  transpose_f32_bf16_kernel<<<dim3(KVB_N / 32, KL / 32), 256, 0, stream>>>(
      w_kvb, wkvb_t, KL, KVB_N);
  transpose_f32_bf16_kernel<<<dim3(H_DIM / 32, (NH * DV) / 32), 256, 0, stream>>>(
      w_o, wo_t, NH * DV, H_DIM);

  // fused = hidden @ w_kv_a  (2048 x 2112 x 2048)
  gemm_bf16_kernel<<<dim3(FUSED_N / 64, S_SEQ / 128), 256, 0, stream>>>(
      hidden_bf, wkva_t, fused, S_SEQ, FUSED_N, H_DIM);

  // rmsnorms
  rmsnorm_bf16_kernel<<<S_SEQ / 8, 256, 0, stream>>>(fused, q_gamma,  qn_bf,  QL, FUSED_N, 0);
  rmsnorm_bf16_kernel<<<S_SEQ / 8, 256, 0, stream>>>(fused, kv_gamma, kvn_bf, KL, FUSED_N, QL);

  // q = qn @ w_qb (2048 x 3072 x 1536); kv = kvn @ w_kvb (2048 x 4096 x 512)
  gemm_bf16_kernel<<<dim3(QB_N / 64, S_SEQ / 128), 256, 0, stream>>>(
      qn_bf, wqb_t, q_f, S_SEQ, QB_N, QL);
  gemm_bf16_kernel<<<dim3(KVB_N / 64, S_SEQ / 128), 256, 0, stream>>>(
      kvn_bf, wkvb_t, kv_f, S_SEQ, KVB_N, KL);

  // rope + assemble
  rope_assemble_kernel<<<(S_SEQ * NH * DQK) / 256, 256, 0, stream>>>(
      q_f, kv_f, fused, qfull, kfull);
  extract_v_t_kernel<<<(S_SEQ * NH * DV) / 256, 256, 0, stream>>>(kv_f, vtg);

  // flash attention
  mla_attn_kernel<<<dim3(S_SEQ / 64, NH), 128, 0, stream>>>(qfull, kfull, vtg, attn_bf);

  // out = attn @ w_o (2048 x 2048 x 2048)
  gemm_bf16_kernel<<<dim3(H_DIM / 64, S_SEQ / 128), 256, 0, stream>>>(
      attn_bf, wo_t, out, S_SEQ, H_DIM, NH * DV);

  (void)in_sizes; (void)n_in; (void)out_size; (void)ws_size;
}
